// Decoder_16149077033073
// MI455X (gfx1250) — matrix operand order repaired and checked by reading, not compiled, not run on hardware
//
#include <hip/hip_runtime.h>
#include <stdint.h>

#define H      1024
#define RH     2048
#define G      (4*RH)       // 8192 gate outputs per layer
#define V      32000
#define BATCH  32
#define TT     128
#define TS     (TT-1)       // 127 decoded steps
#define M_OUT  (TS*BATCH)   // 4064 logits rows
#define SOS_ID 1

typedef __attribute__((ext_vector_type(16))) __bf16        v16bf;
typedef __attribute__((ext_vector_type(8)))  float         v8f;
typedef __attribute__((ext_vector_type(8)))  unsigned short us8;
typedef __attribute__((ext_vector_type(4)))  unsigned short us4;
typedef __attribute__((ext_vector_type(4)))  float         f4;
typedef int v4i_gcc __attribute__((vector_size(16)));      // matches builtin param type

struct Us16 { us8 lo, hi; };   // 32B container -> bit_cast to v16bf

// gfx1250 async global->LDS path (guarded; safe sync fallback otherwise)
#if defined(__gfx1250__) && __has_builtin(__builtin_amdgcn_global_load_async_to_lds_b128)
#define USE_ASYNC_LDS 1
#else
#define USE_ASYNC_LDS 0
#endif

#if USE_ASYNC_LDS
#if __has_builtin(__builtin_amdgcn_s_wait_asynccnt)
#define WAIT_ASYNC(n) __builtin_amdgcn_s_wait_asynccnt(n)
#else
#define WAIT_ASYNC(n) asm volatile("s_wait_asynccnt %0" :: "i"(n))
#endif
#else
#define WAIT_ASYNC(n)
#endif

static __device__ __forceinline__ unsigned short f2bf(float f) {
  unsigned int u = __builtin_bit_cast(unsigned int, f);
  u += 0x7FFFu + ((u >> 16) & 1u);            // round-to-nearest-even
  return (unsigned short)(u >> 16);
}

static __device__ __forceinline__ v8f wmma_bf16(v16bf a, v16bf b, v8f c) {
  return __builtin_amdgcn_wmma_f32_16x16x32_bf16(false, a, false, b, (short)0, c, false, false);
}

// ---------------- weight conversion f32 -> bf16 ----------------
__global__ void cvt_f32_to_bf16(const float* __restrict__ src,
                                unsigned short* __restrict__ dst, long n) {
  long i = ((long)blockIdx.x * blockDim.x + threadIdx.x) * 4;
  long stride = (long)gridDim.x * blockDim.x * 4;
  for (; i < n; i += stride) {
    f4 v = *(const f4*)(src + i);
    us4 o;
    o[0] = f2bf(v[0]); o[1] = f2bf(v[1]); o[2] = f2bf(v[2]); o[3] = f2bf(v[3]);
    *(us4*)(dst + i) = o;
  }
}

// ---------------- embedding gather -> xT[t][k][b] (bf16) ----------------
__global__ void embed_gather(const int* __restrict__ target,
                             const float* __restrict__ emb,
                             unsigned short* __restrict__ XT) {
  long tid = (long)blockIdx.x * blockDim.x + threadIdx.x;
  if (tid >= (long)TS * H * BATCH) return;
  int b = (int)(tid & 31);
  int j = (int)((tid >> 5) & (H - 1));
  int t = (int)(tid >> 15);                 // H*BATCH = 2^15
  int tok = target[b * TT + t];
  XT[tid] = f2bf(emb[(size_t)tok * H + j]); // XT index = (t*H + j)*32 + b == tid
}

// ---------------- initial state: concat even/odd context halves ----------------
__global__ void init_state(const float* __restrict__ ch, const float* __restrict__ cc,
                           unsigned short* __restrict__ hT0, unsigned short* __restrict__ hT1,
                           float* __restrict__ cT0, float* __restrict__ cT1) {
  int tid = blockIdx.x * blockDim.x + threadIdx.x;
  if (tid >= 2 * BATCH * RH) return;
  int b = tid & 31;
  int j = (tid >> 5) & (RH - 1);
  int l = tid >> 16;                        // BATCH*RH = 2^16
  int part = j >> 10;                       // 0: context[2l], 1: context[2l+1]
  int jj = j & (H - 1);
  size_t src = ((size_t)(2 * l + part) * BATCH + b) * H + jj;
  unsigned short* hT = l ? hT1 : hT0;
  float* cT = l ? cT1 : cT0;
  hT[(size_t)j * BATCH + b] = f2bf(ch[src]);   // transposed [k][b] for WMMA B operand
  cT[(size_t)b * RH + j]    = cc[src];
}

// ---------------- fused gate GEMM: gates[b][n] = x@Wih^T + h@Whh^T ----------------
// D = A'(weights 16xK-tile) x B'(activations^T Kx16-tile); 4 waves, 64 n per WG.
// Latency-critical sequential path: no barriers, weights stream from L2 (bf16 LSTM
// weights = 112MB < 192MB L2).
__global__ __launch_bounds__(128) void lstm_gates_wmma(
    const unsigned short* __restrict__ Wih, int K1, const unsigned short* __restrict__ xT,
    const unsigned short* __restrict__ Whh, int K2, const unsigned short* __restrict__ hT,
    float* __restrict__ gates) {
  const int lane = threadIdx.x & 31;
  const int wave = threadIdx.x >> 5;
  const int half = lane >> 4;
  const int col  = lane & 15;
  const int n_base = blockIdx.x * 64 + wave * 16;

  v8f acc0 = {};  // b = 0..15
  v8f acc1 = {};  // b = 16..31
  for (int phase = 0; phase < 2; ++phase) {
    const unsigned short* W  = phase ? Whh : Wih;
    const unsigned short* Bm = phase ? hT  : xT;
    const int K = phase ? K2 : K1;
    const unsigned short* wr = W + (size_t)(n_base + col) * K;
    for (int k0 = 0; k0 < K; k0 += 32) {
      // A fragment: lane holds row n_base+col; K-slots per ISA 16-bit A layout
      us8 c0 = *(const us8*)(wr + k0 + 8 * half);
      us8 c1 = *(const us8*)(wr + k0 + 16 + 8 * half);
      Us16 tmp{c0, c1};
      v16bf a = __builtin_bit_cast(v16bf, tmp);
      // B fragment: lane holds batch column col (b0) / col+16 (b1); K-slots match A
      v16bf b0, b1;
      for (int s = 0; s < 16; ++s) {
        const int kk = k0 + (s < 8 ? 8 * half + s : 16 + 8 * half + (s - 8));
        const unsigned short* bp = Bm + (size_t)kk * BATCH + col;
        b0[s] = __builtin_bit_cast(__bf16, bp[0]);
        b1[s] = __builtin_bit_cast(__bf16, bp[16]);
      }
      if (k0 + 256 < K) __builtin_prefetch(wr + k0 + 256, 0, 1);
      acc0 = wmma_bf16(a, b0, acc0);
      acc1 = wmma_bf16(a, b1, acc1);
    }
  }
  // D[n][b]: VGPR r -> n = n_base + r + 8*half; lane col -> b
  *(v8f*)(gates + (size_t)col        * G + n_base + 8 * half) = acc0;
  *(v8f*)(gates + (size_t)(col + 16) * G + n_base + 8 * half) = acc1;
}

// ---------------- pointwise LSTM cell update ----------------
__global__ void lstm_pointwise(const float* __restrict__ gates,
                               const float* __restrict__ b_ih, const float* __restrict__ b_hh,
                               float* __restrict__ cT, unsigned short* __restrict__ hT,
                               unsigned short* __restrict__ outsT, int mcol) {
  int tid = blockIdx.x * blockDim.x + threadIdx.x;   // BATCH*RH threads
  int j = tid & (RH - 1);
  int b = tid >> 11;
  const float* gb = gates + (size_t)b * G;
  float i_ = gb[j]          + b_ih[j]          + b_hh[j];
  float f_ = gb[RH + j]     + b_ih[RH + j]     + b_hh[RH + j];
  float g_ = gb[2 * RH + j] + b_ih[2 * RH + j] + b_hh[2 * RH + j];
  float o_ = gb[3 * RH + j] + b_ih[3 * RH + j] + b_hh[3 * RH + j];
  float si = 1.f / (1.f + expf(-i_));
  float sf = 1.f / (1.f + expf(-f_));
  float so = 1.f / (1.f + expf(-o_));
  float cn = sf * cT[(size_t)b * RH + j] + si * tanhf(g_);
  float hn = so * tanhf(cn);
  cT[(size_t)b * RH + j] = cn;
  unsigned short hb = f2bf(hn);
  hT[(size_t)j * BATCH + b] = hb;                    // transposed for next GEMM
  if (outsT) outsT[(size_t)j * M_OUT + mcol + b] = hb;
}

// ---------------- output projection: logits -> d_out (bias added) ----------------
// Each WG: 128v x 32m tile, 4 waves (one 32v strip each, 4 accumulators).
// B tile (32k x 32m, 2KB) staged in LDS per WG, double-buffered with
// GLOBAL_LOAD_ASYNC_TO_LDS_B128 + s_wait_asynccnt (sync fallback otherwise).
__global__ __launch_bounds__(128) void out_proj_wmma(
    const unsigned short* __restrict__ Wv, const unsigned short* __restrict__ outsT,
    const float* __restrict__ b_out, float* __restrict__ out) {
  __shared__ __align__(64) unsigned short tile[2][32 * 32];   // 2 x 2KB
  const int tid  = threadIdx.x;
  const int lane = tid & 31;
  const int wave = tid >> 5;
  const int half = lane >> 4;
  const int col  = lane & 15;
  const int v_base = blockIdx.x * 128 + wave * 32;
  const int m_base = blockIdx.y * 32;

  // cooperative fill: 128 threads x 16B = 2KB tile (32 k-rows x 32 m)
  const int frow = tid >> 2;     // 0..31 : k row inside tile
  const int fseg = tid & 3;      // 16B segment along m

  auto fill = [&](int i) {
    const unsigned short* gp =
        outsT + (size_t)(i * 32 + frow) * M_OUT + m_base + fseg * 8;
    unsigned short* lp = &tile[i & 1][frow * 32 + fseg * 8];
#if USE_ASYNC_LDS
    __builtin_amdgcn_global_load_async_to_lds_b128(
        (v4i_gcc*)gp, (v4i_gcc*)lp, 0, 0);
#else
    *(us8*)lp = *(const us8*)gp;
#endif
  };

  v8f acc00 = {}, acc01 = {}, acc10 = {}, acc11 = {};
  const unsigned short* wr0 = Wv + (size_t)(v_base + col) * RH;
  const unsigned short* wr1 = Wv + (size_t)(v_base + 16 + col) * RH;

  fill(0);
  const int NK = RH / 32;   // 64 k-chunks
  for (int i = 0; i < NK; ++i) {
    if (i + 1 < NK) {
      fill(i + 1);
      WAIT_ASYNC(1);        // previous fill (current tile) complete
    } else {
      WAIT_ASYNC(0);        // last tile: everything complete
    }
    __syncthreads();                       // tile[i&1] ready for everyone
    const int k0 = i * 32;
    us8 c0 = *(const us8*)(wr0 + k0 + 8 * half);
    us8 c1 = *(const us8*)(wr0 + k0 + 16 + 8 * half);
    Us16 t0{c0, c1};
    v16bf a0 = __builtin_bit_cast(v16bf, t0);
    us8 d0 = *(const us8*)(wr1 + k0 + 8 * half);
    us8 d1 = *(const us8*)(wr1 + k0 + 16 + 8 * half);
    Us16 t1{d0, d1};
    v16bf a1 = __builtin_bit_cast(v16bf, t1);
    v16bf b0, b1;                          // m col = col (b0) / col+16 (b1)
    for (int s = 0; s < 16; ++s) {
      const int kk = (s < 8 ? 8 * half + s : 16 + 8 * half + (s - 8));
      const unsigned short* bp = &tile[i & 1][kk * 32 + col];
      b0[s] = __builtin_bit_cast(__bf16, bp[0]);
      b1[s] = __builtin_bit_cast(__bf16, bp[16]);
    }
    if (k0 + 256 < RH) {
      __builtin_prefetch(wr0 + k0 + 256, 0, 1);
      __builtin_prefetch(wr1 + k0 + 256, 0, 1);
    }
    acc00 = wmma_bf16(a0, b0, acc00);
    acc01 = wmma_bf16(a0, b1, acc01);
    acc10 = wmma_bf16(a1, b0, acc10);
    acc11 = wmma_bf16(a1, b1, acc11);
    __syncthreads();                       // all reads done before next overwrite
  }

  v8f bias0 = *(const v8f*)(b_out + v_base + 8 * half);
  v8f bias1 = *(const v8f*)(b_out + v_base + 16 + 8 * half);
  acc00 = acc00 + bias0;  acc01 = acc01 + bias0;
  acc10 = acc10 + bias1;  acc11 = acc11 + bias1;
  const int m0 = m_base + col, m1 = m0 + 16;
  const int t0i = m0 >> 5, b0i = m0 & 31;
  const int t1i = m1 >> 5, b1i = m1 & 31;
  *(v8f*)(out + ((size_t)b0i * TT + t0i + 1) * V + v_base + 8 * half)      = acc00;
  *(v8f*)(out + ((size_t)b1i * TT + t1i + 1) * V + v_base + 8 * half)      = acc01;
  *(v8f*)(out + ((size_t)b0i * TT + t0i + 1) * V + v_base + 16 + 8 * half) = acc10;
  *(v8f*)(out + ((size_t)b1i * TT + t1i + 1) * V + v_base + 16 + 8 * half) = acc11;
}

// ---------------- t=0 one-hot SOS rows ----------------
__global__ void write_sos(float* __restrict__ out) {
  long tid = (long)blockIdx.x * blockDim.x + threadIdx.x;
  if (tid >= (long)BATCH * V) return;
  int b = (int)(tid / V);
  int v = (int)(tid - (long)b * V);
  out[(size_t)b * TT * V + v] = (v == SOS_ID) ? 1.0f : 0.0f;
}

// ---------------- in-place log-softmax over V for each (b, t>=1) row ----------------
__global__ __launch_bounds__(256) void log_softmax_inplace(float* __restrict__ out) {
  __shared__ float red[256];
  int m = blockIdx.x;                 // 0..M_OUT-1
  int t = m >> 5, b = m & 31;
  float* row = out + ((size_t)b * TT + t + 1) * V;
  int tid = threadIdx.x;
  float mx = -3.4e38f;
  for (int v = tid; v < V; v += 256) mx = fmaxf(mx, row[v]);
  red[tid] = mx; __syncthreads();
  for (int s = 128; s > 0; s >>= 1) { if (tid < s) red[tid] = fmaxf(red[tid], red[tid + s]); __syncthreads(); }
  mx = red[0]; __syncthreads();
  float sum = 0.f;
  for (int v = tid; v < V; v += 256) sum += expf(row[v] - mx);
  red[tid] = sum; __syncthreads();
  for (int s = 128; s > 0; s >>= 1) { if (tid < s) red[tid] += red[tid + s]; __syncthreads(); }
  float lse = mx + logf(red[0]);
  for (int v = tid; v < V; v += 256) row[v] = row[v] - lse;
}

extern "C" void kernel_launch(void* const* d_in, const int* in_sizes, int n_in,
                              void* d_out, int out_size, void* d_ws, size_t ws_size,
                              hipStream_t stream) {
  const int*   target = (const int*)d_in[1];
  const float* ctx_h  = (const float*)d_in[2];
  const float* ctx_c  = (const float*)d_in[3];
  const float* emb    = (const float*)d_in[4];
  const float* Wih0   = (const float*)d_in[5];
  const float* Whh0   = (const float*)d_in[6];
  const float* bih0   = (const float*)d_in[7];
  const float* bhh0   = (const float*)d_in[8];
  const float* Wih1   = (const float*)d_in[9];
  const float* Whh1   = (const float*)d_in[10];
  const float* bih1   = (const float*)d_in[11];
  const float* bhh1   = (const float*)d_in[12];
  const float* Wout   = (const float*)d_in[13];
  const float* bout   = (const float*)d_in[14];
  float* out = (float*)d_out;

  uint8_t* ws = (uint8_t*)d_ws;
  size_t off = 0;
  auto walloc = [&](size_t bytes) -> void* {
    void* p = ws + off;
    off += (bytes + 255) & ~(size_t)255;
    return p;
  };
  unsigned short* Wih0b = (unsigned short*)walloc((size_t)G * H  * 2);
  unsigned short* Whh0b = (unsigned short*)walloc((size_t)G * RH * 2);
  unsigned short* Wih1b = (unsigned short*)walloc((size_t)G * RH * 2);
  unsigned short* Whh1b = (unsigned short*)walloc((size_t)G * RH * 2);
  unsigned short* Woutb = (unsigned short*)walloc((size_t)V * RH * 2);
  unsigned short* XT    = (unsigned short*)walloc((size_t)TS * H * BATCH * 2);
  unsigned short* outsT = (unsigned short*)walloc((size_t)RH * M_OUT * 2);
  unsigned short* hT0   = (unsigned short*)walloc((size_t)RH * BATCH * 2);
  unsigned short* hT1   = (unsigned short*)walloc((size_t)RH * BATCH * 2);
  float*          cT0   = (float*)walloc((size_t)BATCH * RH * 4);
  float*          cT1   = (float*)walloc((size_t)BATCH * RH * 4);
  float*          gates = (float*)walloc((size_t)BATCH * G * 4);

  // 1) bf16 weight images (LSTM weights then fit in the 192MB L2 -> recurrence runs out of L2)
  cvt_f32_to_bf16<<<1024, 256, 0, stream>>>(Wih0, Wih0b, (long)G * H);
  cvt_f32_to_bf16<<<1024, 256, 0, stream>>>(Whh0, Whh0b, (long)G * RH);
  cvt_f32_to_bf16<<<1024, 256, 0, stream>>>(Wih1, Wih1b, (long)G * RH);
  cvt_f32_to_bf16<<<1024, 256, 0, stream>>>(Whh1, Whh1b, (long)G * RH);
  cvt_f32_to_bf16<<<1024, 256, 0, stream>>>(Wout, Woutb, (long)V * RH);

  // 2) inputs & state
  {
    long n = (long)TS * H * BATCH;
    embed_gather<<<(unsigned)((n + 255) / 256), 256, 0, stream>>>(target, emb, XT);
  }
  init_state<<<(2 * BATCH * RH) / 256, 256, 0, stream>>>(ctx_h, ctx_c, hT0, hT1, cT0, cT1);

  // 3) recurrence: 127 steps x 2 layers (sequential by construction)
  for (int t = 0; t < TS; ++t) {
    const unsigned short* xt = XT + (size_t)t * H * BATCH;
    lstm_gates_wmma<<<G / 64, 128, 0, stream>>>(Wih0b, H,  xt,  Whh0b, RH, hT0, gates);
    lstm_pointwise<<<(BATCH * RH) / 256, 256, 0, stream>>>(gates, bih0, bhh0, cT0, hT0, nullptr, 0);
    lstm_gates_wmma<<<G / 64, 128, 0, stream>>>(Wih1b, RH, hT0, Whh1b, RH, hT1, gates);
    lstm_pointwise<<<(BATCH * RH) / 256, 256, 0, stream>>>(gates, bih1, bhh1, cT1, hT1, outsT, t * BATCH);
  }

  // 4) big output GEMM (532 GFLOP, bf16 WMMA, LDS-staged B, 128v x 32m per WG)
  out_proj_wmma<<<dim3(V / 128, M_OUT / 32), 128, 0, stream>>>(Woutb, outsT, bout, out);

  // 5) t=0 one-hot rows and in-place log-softmax of the rest
  write_sos<<<(unsigned)(((long)BATCH * V + 255) / 256), 256, 0, stream>>>(out);
  log_softmax_inplace<<<M_OUT, 256, 0, stream>>>(out);
}
